// ProbAttention_4492535792233
// MI455X (gfx1250) — matrix operand order repaired and checked by reading, not compiled, not run on hardware
//
#include <hip/hip_runtime.h>
#include <math.h>

typedef _Float16 v16h __attribute__((ext_vector_type(16)));
typedef _Float16 v8h  __attribute__((ext_vector_type(8)));
typedef _Float16 v4h  __attribute__((ext_vector_type(4)));
typedef float    v8f  __attribute__((ext_vector_type(8)));

#define L_DIM    4096
#define E_DIM    64
#define NCHUNK   16
#define CHUNKLEN 256
#define BK       64      // keys per block-iteration in k_attn
#define ST       72      // padded f16 row stride for LDS tiles

// ---------------------------------------------------------------------------
// Kernel 1: sparse sampled scores  M[b,l] = max_s(q.k_s) - mean over L
// ---------------------------------------------------------------------------
__global__ __launch_bounds__(256) void k_sample_scores(
    const float* __restrict__ Q, const float* __restrict__ K,
    const int* __restrict__ samp, float* __restrict__ M, int SK, int total)
{
    int idx = blockIdx.x * 256 + threadIdx.x;
    if (idx >= total) return;
    int l = idx & (L_DIM - 1);
    int b = idx >> 12;

    const float4* q4 = (const float4*)(Q + (size_t)idx * E_DIM);
    float4 q[16];
#pragma unroll
    for (int j = 0; j < 16; ++j) q[j] = q4[j];

    const int* sp = samp + (size_t)l * SK;
    float mx = -3.4e38f, sm = 0.f;
    for (int s = 0; s < SK; ++s) {
        int ki = sp[s];
        const float4* kp = (const float4*)(K + ((size_t)b * L_DIM + ki) * E_DIM);
        float dot = 0.f;
#pragma unroll
        for (int j = 0; j < 16; ++j) {
            float4 kv = kp[j];
            dot += q[j].x * kv.x + q[j].y * kv.y + q[j].z * kv.z + q[j].w * kv.w;
        }
        mx = fmaxf(mx, dot);
        sm += dot;
    }
    M[idx] = mx - sm * (1.0f / (float)L_DIM);
}

// ---------------------------------------------------------------------------
// Kernel 2: per-batch top-u (iterative argmax, ties -> smallest index)
// ---------------------------------------------------------------------------
__global__ __launch_bounds__(256) void k_topk(
    const float* __restrict__ M, int* __restrict__ Mtop, int u)
{
    __shared__ float vals[L_DIM];
    __shared__ float rv[256];
    __shared__ int   ri[256];
    int b = blockIdx.x, tid = threadIdx.x;

    for (int i = tid; i < L_DIM; i += 256) vals[i] = M[(size_t)b * L_DIM + i];
    __syncthreads();

    for (int it = 0; it < u; ++it) {
        float best = -3.4e38f; int bidx = L_DIM;
        for (int i = tid; i < L_DIM; i += 256) {
            float v = vals[i];
            if (v > best || (v == best && i < bidx)) { best = v; bidx = i; }
        }
        rv[tid] = best; ri[tid] = bidx;
        __syncthreads();
        for (int s = 128; s > 0; s >>= 1) {
            if (tid < s) {
                if (rv[tid + s] > rv[tid] ||
                    (rv[tid + s] == rv[tid] && ri[tid + s] < ri[tid])) {
                    rv[tid] = rv[tid + s]; ri[tid] = ri[tid + s];
                }
            }
            __syncthreads();
        }
        if (tid == 0) { Mtop[(size_t)b * u + it] = ri[0]; vals[ri[0]] = -3.4e38f; }
        __syncthreads();
    }
}

// ---------------------------------------------------------------------------
// Kernel 3a/3b/3c: chunked cumsum of V along L (per batch, per column)
// ---------------------------------------------------------------------------
__global__ __launch_bounds__(64) void k_cumsum_local(
    const float* __restrict__ V, float* __restrict__ OUT, float* __restrict__ sums)
{
    int b = blockIdx.x >> 4, c = blockIdx.x & 15, e = threadIdx.x;
    size_t base = ((size_t)b * L_DIM + c * CHUNKLEN) * E_DIM + e;
    float acc = 0.f;
    for (int i = 0; i < CHUNKLEN; ++i) {
        acc += V[base + (size_t)i * E_DIM];
        OUT[base + (size_t)i * E_DIM] = acc;
    }
    sums[(size_t)blockIdx.x * E_DIM + e] = acc;
}

__global__ __launch_bounds__(64) void k_cumsum_prefix(float* __restrict__ sums)
{
    int b = blockIdx.x, e = threadIdx.x;
    float run = 0.f;
    for (int c = 0; c < NCHUNK; ++c) {
        size_t idx = ((size_t)b * NCHUNK + c) * E_DIM + e;
        float t = sums[idx]; sums[idx] = run; run += t;
    }
}

__global__ __launch_bounds__(64) void k_cumsum_fix(
    float* __restrict__ OUT, const float* __restrict__ sums)
{
    int b = blockIdx.x >> 4, c = blockIdx.x & 15, e = threadIdx.x;
    if (c == 0) return;
    float off = sums[(size_t)blockIdx.x * E_DIM + e];
    size_t base = ((size_t)b * L_DIM + c * CHUNKLEN) * E_DIM + e;
    for (int i = 0; i < CHUNKLEN; ++i) OUT[base + (size_t)i * E_DIM] += off;
}

// ---------------------------------------------------------------------------
// Kernel 4: dense attention for the selected queries (WMMA f16, f32 accum).
// One block per batch; each wave owns one 16-row tile of the u queries.
// 64-key block iterations; K and V tiles staged once per block in LDS (f16),
// prefetch of the next tile; deferred (per-lane) softmax normalizer.
// ---------------------------------------------------------------------------
__global__ __launch_bounds__(128) void k_attn(
    const float* __restrict__ Q, const float* __restrict__ K,
    const float* __restrict__ V, const int* __restrict__ Mtop,
    float* __restrict__ OUT, int u)
{
    __shared__ _Float16 Kh[BK * ST];        // [k_local][e]        (B frags of QK^T)
    __shared__ _Float16 VT[E_DIM * ST];     // [e][k_local]        (B frags of PV)
    __shared__ _Float16 PT[4 * 16 * ST];    // per-wave P tiles [m][k_local]

    const int tid   = threadIdx.x;
    const int wv    = tid >> 5;
    const int lane  = tid & 31;
    const int hlane = lane & 15;
    const bool hiH  = lane >= 16;
    const int koff  = hiH ? 8  : 0;         // A-layout K sub-offset
    const int koff2 = hiH ? 16 : 0;         // contiguous-order K sub-offset (unused)
    const int b     = blockIdx.x;
    const int nthr  = blockDim.x;

    // ---- load Q fragments (A layout, pre-scaled by 1/sqrt(E), f16) ----
    int mg   = wv * 16 + hlane;
    int qidx = Mtop[b * u + (mg < u ? mg : u - 1)];
    const float* qrow = Q + ((size_t)b * L_DIM + qidx) * E_DIM;
    const float scale = 0.125f;
    v16h aq[2];
#pragma unroll
    for (int h = 0; h < 2; ++h) {
        int k0 = h * 32;
        const float4* qa = (const float4*)(qrow + k0 + koff);
        const float4* qb = (const float4*)(qrow + k0 + 16 + koff);
        float4 t0 = qa[0], t1 = qa[1], t2 = qb[0], t3 = qb[1];
        aq[h][0]  = (_Float16)(t0.x * scale); aq[h][1]  = (_Float16)(t0.y * scale);
        aq[h][2]  = (_Float16)(t0.z * scale); aq[h][3]  = (_Float16)(t0.w * scale);
        aq[h][4]  = (_Float16)(t1.x * scale); aq[h][5]  = (_Float16)(t1.y * scale);
        aq[h][6]  = (_Float16)(t1.z * scale); aq[h][7]  = (_Float16)(t1.w * scale);
        aq[h][8]  = (_Float16)(t2.x * scale); aq[h][9]  = (_Float16)(t2.y * scale);
        aq[h][10] = (_Float16)(t2.z * scale); aq[h][11] = (_Float16)(t2.w * scale);
        aq[h][12] = (_Float16)(t3.x * scale); aq[h][13] = (_Float16)(t3.y * scale);
        aq[h][14] = (_Float16)(t3.z * scale); aq[h][15] = (_Float16)(t3.w * scale);
    }

    v8f o[4] = {};
    float mrun[8], lpart[8];                // lpart: per-lane partial normalizer
#pragma unroll
    for (int r = 0; r < 8; ++r) { mrun[r] = -3.0e38f; lpart[r] = 0.f; }

    _Float16* myP = &PT[wv * 16 * ST];

    for (int key0 = 0; key0 < L_DIM; key0 += BK) {
        __syncthreads();
        // ---- stage K tile (row-major f16) and V tile (transposed f16) ----
        for (int j = tid; j < BK * 16; j += nthr) {
            int k = j >> 4, e4 = (j & 15) << 2;
            float4 t = ((const float4*)(K + ((size_t)b * L_DIM + key0 + k) * E_DIM))[j & 15];
            v4h p;
            p[0] = (_Float16)t.x; p[1] = (_Float16)t.y;
            p[2] = (_Float16)t.z; p[3] = (_Float16)t.w;
            *(v4h*)&Kh[k * ST + e4] = p;
        }
        for (int j = tid; j < BK * 16; j += nthr) {
            int k = j >> 4, e4 = (j & 15) << 2;
            float4 t = ((const float4*)(V + ((size_t)b * L_DIM + key0 + k) * E_DIM))[j & 15];
            VT[(e4 + 0) * ST + k] = (_Float16)t.x;
            VT[(e4 + 1) * ST + k] = (_Float16)t.y;
            VT[(e4 + 2) * ST + k] = (_Float16)t.z;
            VT[(e4 + 3) * ST + k] = (_Float16)t.w;
        }
        // ---- prefetch next tile into cache while we compute this one ----
        if (key0 + BK < L_DIM) {
            const char* nk = (const char*)(K + ((size_t)b * L_DIM + key0 + BK) * E_DIM);
            const char* nv = (const char*)(V + ((size_t)b * L_DIM + key0 + BK) * E_DIM);
            for (int j = tid; j < 128; j += nthr) {
                __builtin_prefetch(nk + j * 128, 0, 3);
                __builtin_prefetch(nv + j * 128, 0, 3);
            }
        }
        __syncthreads();

        // ---- QK^T over 64 keys: four 16-key n-groups, K=64 in 2 WMMA steps ----
        v8f s[4] = {};
#pragma unroll
        for (int g = 0; g < 4; ++g) {
            int n = g * 16 + hlane;
#pragma unroll
            for (int h = 0; h < 2; ++h) {
                int k0 = h * 32;
                v8h lo = *(const v8h*)&Kh[n * ST + k0 + koff];
                v8h hi = *(const v8h*)&Kh[n * ST + k0 + 16 + koff];
                v16h bk = __builtin_shufflevector(lo, hi, 0,1,2,3,4,5,6,7,8,9,10,11,12,13,14,15);
                s[g] = __builtin_amdgcn_wmma_f32_16x16x32_f16(false, aq[h], false, bk,
                                                              (short)0, s[g], false, false);
            }
        }

        // ---- streaming softmax (row max reduced; normalizer kept per-lane) ----
        float corr[8];
#pragma unroll
        for (int r = 0; r < 8; ++r) {
            float v = fmaxf(fmaxf(s[0][r], s[1][r]), fmaxf(s[2][r], s[3][r]));
            v = fmaxf(v, __shfl_xor(v, 1, 32));
            v = fmaxf(v, __shfl_xor(v, 2, 32));
            v = fmaxf(v, __shfl_xor(v, 4, 32));
            v = fmaxf(v, __shfl_xor(v, 8, 32));
            float mnew = fmaxf(mrun[r], v);
            float p0 = __expf(s[0][r] - mnew);
            float p1 = __expf(s[1][r] - mnew);
            float p2 = __expf(s[2][r] - mnew);
            float p3 = __expf(s[3][r] - mnew);
            float c = __expf(mrun[r] - mnew);
            lpart[r] = lpart[r] * c + (p0 + p1) + (p2 + p3);
            mrun[r] = mnew;
            corr[r] = c;
            int m = r + (hiH ? 8 : 0);
            myP[m * ST + hlane]      = (_Float16)p0;   // D-layout -> LDS
            myP[m * ST + 16 + hlane] = (_Float16)p1;
            myP[m * ST + 32 + hlane] = (_Float16)p2;
            myP[m * ST + 48 + hlane] = (_Float16)p3;
        }
#pragma unroll
        for (int r = 0; r < 8; ++r) {
            o[0][r] *= corr[r]; o[1][r] *= corr[r];
            o[2][r] *= corr[r]; o[3][r] *= corr[r];
        }

        asm volatile("s_wait_dscnt 0" ::: "memory");   // P store -> reload (same wave)

        // ---- re-read P in A layout (two 16x32 fragments) ----
        v16h a2[2];
#pragma unroll
        for (int h = 0; h < 2; ++h) {
            int k0 = h * 32;
            v8h lo = *(const v8h*)&myP[hlane * ST + k0 + koff];
            v8h hi = *(const v8h*)&myP[hlane * ST + k0 + 16 + koff];
            a2[h] = __builtin_shufflevector(lo, hi, 0,1,2,3,4,5,6,7,8,9,10,11,12,13,14,15);
        }
        // ---- PV: 4 e-groups x 2 K-steps ----
#pragma unroll
        for (int g = 0; g < 4; ++g) {
            int e = g * 16 + hlane;
#pragma unroll
            for (int h = 0; h < 2; ++h) {
                int k0 = h * 32;
                v8h lo = *(const v8h*)&VT[e * ST + k0 + koff];
                v8h hi = *(const v8h*)&VT[e * ST + k0 + 16 + koff];
                v16h bv = __builtin_shufflevector(lo, hi, 0,1,2,3,4,5,6,7,8,9,10,11,12,13,14,15);
                o[g] = __builtin_amdgcn_wmma_f32_16x16x32_f16(false, a2[h], false, bv,
                                                              (short)0, o[g], false, false);
            }
        }
    }

    // ---- final normalizer reduction, normalize, scatter selected rows ----
#pragma unroll
    for (int r = 0; r < 8; ++r) {
        float l = lpart[r];
        l += __shfl_xor(l, 1, 32);
        l += __shfl_xor(l, 2, 32);
        l += __shfl_xor(l, 4, 32);
        l += __shfl_xor(l, 8, 32);
        int m = r + (hiH ? 8 : 0);
        int mgr = wv * 16 + m;
        if (mgr < u) {
            int row = Mtop[b * u + mgr];
            float inv = 1.0f / l;
            float* op = OUT + ((size_t)b * L_DIM + row) * E_DIM + hlane;
            op[0]  = o[0][r] * inv;
            op[16] = o[1][r] * inv;
            op[32] = o[2][r] * inv;
            op[48] = o[3][r] * inv;
        }
    }
}

// ---------------------------------------------------------------------------
extern "C" void kernel_launch(void* const* d_in, const int* in_sizes, int n_in,
                              void* d_out, int out_size, void* d_ws, size_t ws_size,
                              hipStream_t stream)
{
    const float* Q    = (const float*)d_in[0];
    const float* K    = (const float*)d_in[1];
    const float* V    = (const float*)d_in[2];
    const int*   samp = (const int*)d_in[3];
    float*       OUT  = (float*)d_out;

    const int B  = in_sizes[0] / (L_DIM * E_DIM);
    const int SK = in_sizes[3] / L_DIM;    // = 45
    const int u  = SK;                     // = 45

    size_t offM    = 0;
    size_t offTop  = offM + (size_t)B * L_DIM * sizeof(float);
    size_t offSums = (offTop + (size_t)B * u * sizeof(int) + 255) & ~(size_t)255;

    float* M    = (float*)((char*)d_ws + offM);
    int*   Mtop = (int*)((char*)d_ws + offTop);
    float* sums = (float*)((char*)d_ws + offSums);

    int total = B * L_DIM;
    k_sample_scores<<<(total + 255) / 256, 256, 0, stream>>>(Q, K, samp, M, SK, total);
    k_topk<<<B, 256, 0, stream>>>(M, Mtop, u);
    k_cumsum_local<<<B * NCHUNK, 64, 0, stream>>>(V, OUT, sums);
    k_cumsum_prefix<<<B, 64, 0, stream>>>(sums);
    k_cumsum_fix<<<B * NCHUNK, 64, 0, stream>>>(OUT, sums);

    int ntiles = (u + 15) / 16;            // = 3
    if (ntiles > 4) ntiles = 4;            // LDS sized for up to 4 waves
    k_attn<<<B, 32 * ntiles, 0, stream>>>(Q, K, V, Mtop, OUT, u);
}
